// TransitionUp_29472065585605
// MI455X (gfx1250) — matrix operand order repaired and checked by reading, not compiled, not run on hardware
//
#include <hip/hip_runtime.h>

typedef __attribute__((ext_vector_type(16))) _Float16 v16h;
typedef __attribute__((ext_vector_type(8)))  _Float16 v8h;
typedef __attribute__((ext_vector_type(8)))  float    v8f;

#define LN_EPS 1e-5f
#define WPAD   8      // LDS row padding in halves: +16B -> +4 banks per row

// CDNA5 async copy: 16B global -> LDS, tracked by ASYNCcnt.
// GVS form: vdst = LDS byte offset (VGPR), vaddr = 32-bit global byte offset
// (VGPR), saddr = 64-bit base (SGPR pair).
__device__ __forceinline__ void async_g2l_b128(unsigned lds_off, unsigned g_off,
                                               const void* g_base) {
    asm volatile("global_load_async_to_lds_b128 %0, %1, %2"
                 :: "v"(lds_off), "v"(g_off), "s"(g_base) : "memory");
}
__device__ __forceinline__ void wait_async0() {
    asm volatile("s_wait_asynccnt 0x0" ::: "memory");
}

// ---- A fragment (16x32 f16, CDNA5 layout) from an fp32 row ----
// lane = kh*16 + m ; elem i<8 -> K = kc*32 + kh*8 + i
//                    elem i>=8 -> K = kc*32 + 16 + kh*8 + (i-8)
__device__ __forceinline__ v16h make_a_frag(const float* rowp, int kh, int kc) {
    const float* p0 = rowp + kc * 32 + kh * 8;
    const float* p1 = p0 + 16;
    v16h a;
#pragma unroll
    for (int i = 0; i < 8; ++i) a[i] = (_Float16)p0[i];
#pragma unroll
    for (int i = 0; i < 8; ++i) a[8 + i] = (_Float16)p1[i];
    return a;
}

// ---- B fragment (32x16 f16) from LDS (n-major, padded row stride ldK) ----
// lane = kh*16 + n ; elem i<8 -> K = kh*8 + i ; elem i>=8 -> K = 16 + kh*8 + (i-8)
__device__ __forceinline__ v16h load_b_frag_lds(const _Float16* w, int ldK,
                                                int t, int kc, int lane) {
    const int n  = t * 16 + (lane & 15);
    const int k0 = kc * 32 + (lane >> 4) * 8;
    const _Float16* p = w + n * ldK + k0;         // 16B-aligned
    v8h lo = *(const v8h*)p;
    v8h hi = *(const v8h*)(p + 16);
    v16h b;
#pragma unroll
    for (int i = 0; i < 8; ++i) { b[i] = lo[i]; b[8 + i] = hi[i]; }
    return b;
}

// Fused Linear -> LN -> Linear -> LN -> ReLU over D=128 outputs.
// 8 waves / block, each wave owns 16 rows; weights async-staged to LDS.
template <int K1>
__global__ __launch_bounds__(256) void mlp_ln_kernel(
    const float* __restrict__ x, const _Float16* __restrict__ wT1,
    const float* __restrict__ g1, const float* __restrict__ b1,
    const _Float16* __restrict__ wT2,
    const float* __restrict__ g2, const float* __restrict__ b2,
    float* __restrict__ out0, float* __restrict__ out1)
{
    constexpr int D    = 128;
    constexpr int KC1  = K1 / 32;
    constexpr int LD1  = K1 + WPAD;
    constexpr int LD2  = D + WPAD;

    __shared__ __align__(16) _Float16 w1s[D * LD1];
    __shared__ __align__(16) _Float16 w2s[D * LD2];
    __shared__ float slab_all[8 * 16 * D];        // 64 KB

    // ---------- async-stage weights into LDS (whole block) ----------
    {
        const unsigned l1 = (unsigned)(size_t)(void*)w1s;
        const unsigned l2 = (unsigned)(size_t)(void*)w2s;
        constexpr int C1 = K1 / 8;                // 16B chunks per row
        constexpr int C2 = D / 8;
        for (int i = threadIdx.x; i < D * C1; i += 256) {
            const int n = i / C1, c = i - n * C1;
            async_g2l_b128(l1 + (unsigned)(n * LD1 + c * 8) * 2u,
                           (unsigned)(n * K1 + c * 8) * 2u, wT1);
        }
        for (int i = threadIdx.x; i < D * C2; i += 256) {
            const int n = i / C2, c = i - n * C2;
            async_g2l_b128(l2 + (unsigned)(n * LD2 + c * 8) * 2u,
                           (unsigned)(n * D + c * 8) * 2u, wT2);
        }
        wait_async0();
    }
    __syncthreads();

    const int wave = threadIdx.x >> 5;
    const int lane = threadIdx.x & 31;
    const int m    = lane & 15;       // row within wave slab
    const int kh   = lane >> 4;       // K-half / column-half selector
    float* slab = slab_all + wave * 16 * D;
    const long rowBase = (long)blockIdx.x * 128 + wave * 16;

    // ---------------- GEMM1: h = x @ w1 ----------------
    const float* xrow = x + (rowBase + m) * (long)K1;
    v16h a1[KC1];
#pragma unroll
    for (int kc = 0; kc < KC1; ++kc) a1[kc] = make_a_frag(xrow, kh, kc);

#pragma unroll
    for (int t = 0; t < 8; ++t) {
        v16h bf[KC1];
#pragma unroll
        for (int kc = 0; kc < KC1; ++kc) bf[kc] = load_b_frag_lds(w1s, LD1, t, kc, lane);
        v8f acc = {};
#pragma unroll
        for (int kc = 0; kc < KC1; ++kc)
            acc = __builtin_amdgcn_wmma_f32_16x16x32_f16(
                false, a1[kc], false, bf[kc], (short)0, acc, false, false);
        // C layout: VGPR r holds M = r + 8*kh, N = t*16 + m
#pragma unroll
        for (int r = 0; r < 8; ++r)
            slab[(r + kh * 8) * D + t * 16 + m] = acc[r];
    }
    __syncthreads();

    // ---------------- LayerNorm 1 (2 lanes per row) ----------------
    {
        float* row = slab + m * D;
        float s1 = 0.f, s2 = 0.f;
#pragma unroll 8
        for (int i = 0; i < 64; ++i) {
            float v = row[kh * 64 + i];
            s1 += v; s2 += v * v;
        }
        s1 += __shfl_xor(s1, 16, 32);
        s2 += __shfl_xor(s2, 16, 32);
        const float mean = s1 * (1.f / 128.f);
        const float var  = s2 * (1.f / 128.f) - mean * mean;
        const float rstd = rsqrtf(var + LN_EPS);
#pragma unroll 8
        for (int i = 0; i < 64; ++i) {
            const int c = kh * 64 + i;
            row[c] = (row[c] - mean) * rstd * g1[c] + b1[c];
        }
    }
    __syncthreads();

    // ---------------- GEMM2: h2 = LN1(h) @ w2 ----------------
    // Preload A fragments before the tile loop overwrites the slab
    // (LDS ops issue in order within a wave, so loads precede stores).
    v16h a2[4];
#pragma unroll
    for (int kc = 0; kc < 4; ++kc) a2[kc] = make_a_frag(slab + m * D, kh, kc);

#pragma unroll
    for (int t = 0; t < 8; ++t) {
        v16h bf[4];
#pragma unroll
        for (int kc = 0; kc < 4; ++kc) bf[kc] = load_b_frag_lds(w2s, LD2, t, kc, lane);
        v8f acc = {};
#pragma unroll
        for (int kc = 0; kc < 4; ++kc)
            acc = __builtin_amdgcn_wmma_f32_16x16x32_f16(
                false, a2[kc], false, bf[kc], (short)0, acc, false, false);
#pragma unroll
        for (int r = 0; r < 8; ++r)
            slab[(r + kh * 8) * D + t * 16 + m] = acc[r];
    }
    __syncthreads();

    // ---------------- LayerNorm 2 + ReLU + store ----------------
    {
        float* row = slab + m * D;
        float s1 = 0.f, s2 = 0.f;
#pragma unroll 8
        for (int i = 0; i < 64; ++i) {
            float v = row[kh * 64 + i];
            s1 += v; s2 += v * v;
        }
        s1 += __shfl_xor(s1, 16, 32);
        s2 += __shfl_xor(s2, 16, 32);
        const float mean = s1 * (1.f / 128.f);
        const float var  = s2 * (1.f / 128.f) - mean * mean;
        const float rstd = rsqrtf(var + LN_EPS);
        float* d0 = out0 + (rowBase + m) * (long)D;
        float* d1 = out1 ? out1 + (rowBase + m) * (long)D : nullptr;
#pragma unroll 8
        for (int i = 0; i < 64; ++i) {
            const int c = kh * 64 + i;
            float v = (row[c] - mean) * rstd * g2[c] + b2[c];
            v = fmaxf(v, 0.f);
            d0[c] = v;
            if (d1) d1[c] = v;
        }
    }
}

// Transpose weights to n-major f16 so staging/fragments are contiguous 16B+.
__global__ __launch_bounds__(256) void prep_weights_kernel(
    const float* __restrict__ w1c, const float* __restrict__ w2c,
    const float* __restrict__ w1f, const float* __restrict__ w2f,
    _Float16* __restrict__ t1c, _Float16* __restrict__ t2c,
    _Float16* __restrict__ t1f, _Float16* __restrict__ t2f)
{
    const int stride = gridDim.x * blockDim.x;
    const int tid = blockIdx.x * blockDim.x + threadIdx.x;
    for (int i = tid; i < 256 * 128; i += stride) {       // w1c: 256 x 128
        const int k = i >> 7, n = i & 127;
        t1c[n * 256 + k] = (_Float16)w1c[i];
    }
    for (int i = tid; i < 128 * 128; i += stride) {       // three 128 x 128
        const int k = i >> 7, n = i & 127;
        t2c[n * 128 + k] = (_Float16)w2c[i];
        t1f[n * 128 + k] = (_Float16)w1f[i];
        t2f[n * 128 + k] = (_Float16)w2f[i];
    }
}

// out[nr,nc,:] = fc[entry % (Lc*N), :] + ff_orig[nr,nc,:]
// One wave (32 lanes x float4) per entry; gathers from the pre-scatter
// snapshot so duplicate targets behave like the reference's gather-then-set.
__global__ __launch_bounds__(256) void scatter_add_kernel(
    const float* __restrict__ fc, const float* __restrict__ ff_orig,
    const int* __restrict__ nrow, const int* __restrict__ ncol,
    float* __restrict__ out, int total, int LcN, int N)
{
    const long gid = (long)blockIdx.x * blockDim.x + threadIdx.x;
    const int entry = (int)(gid >> 5);
    const int l = (int)(gid & 31);
    if (entry >= total) return;
    const int r = nrow[entry];
    const int c = ncol[entry];
    const int rest = entry % LcN;
    const float4* sfc = (const float4*)(fc + (long)rest * 128);
    const float4* sff = (const float4*)(ff_orig + ((long)r * N + c) * 128);
    float4* dst = (float4*)(out + ((long)r * N + c) * 128);
    const float4 a = sfc[l];
    const float4 b = sff[l];
    dst[l] = make_float4(a.x + b.x, a.y + b.y, a.z + b.z, a.w + b.w);
}

extern "C" void kernel_launch(void* const* d_in, const int* in_sizes, int n_in,
                              void* d_out, int out_size, void* d_ws, size_t ws_size,
                              hipStream_t stream) {
    const float* fcoarse = (const float*)d_in[0];   // (1024, 32, 256)
    const float* ffine   = (const float*)d_in[1];   // (4096, 32, 128)
    const float* w1c = (const float*)d_in[2];
    const float* g1c = (const float*)d_in[3];
    const float* b1c = (const float*)d_in[4];
    const float* w2c = (const float*)d_in[5];
    const float* g2c = (const float*)d_in[6];
    const float* b2c = (const float*)d_in[7];
    const float* w1f = (const float*)d_in[8];
    const float* g1f = (const float*)d_in[9];
    const float* b1f = (const float*)d_in[10];
    const float* w2f = (const float*)d_in[11];
    const float* g2f = (const float*)d_in[12];
    const float* b2f = (const float*)d_in[13];
    const int* nrow  = (const int*)d_in[14];
    const int* ncol  = (const int*)d_in[15];
    // d_in[16] = keep_idx: unused by the reference output.

    const int Lc = 1024, N = 32, D = 128;
    const int rowsC = Lc * N;                 // 32768
    const int rowsF = in_sizes[1] / D;        // 131072
    const int total = in_sizes[14];           // K*Lc*N = 262144

    // Workspace layout: f16 weights (160KB) | fc (16MB) | ff snapshot (64MB)
    char* ws = (char*)d_ws;
    _Float16* t1c = (_Float16*)(ws);
    _Float16* t2c = (_Float16*)(ws + (64 << 10));
    _Float16* t1f = (_Float16*)(ws + (96 << 10));
    _Float16* t2f = (_Float16*)(ws + (128 << 10));
    float* fcBuf  = (float*)(ws + (160 << 10));
    float* ffOrig = (float*)(ws + (160 << 10) + (size_t)rowsC * D * sizeof(float));

    prep_weights_kernel<<<64, 256, 0, stream>>>(w1c, w2c, w1f, w2f, t1c, t2c, t1f, t2f);

    mlp_ln_kernel<256><<<rowsC / 128, 256, 0, stream>>>(
        fcoarse, t1c, g1c, b1c, t2c, g2c, b2c, fcBuf, nullptr);

    mlp_ln_kernel<128><<<rowsF / 128, 256, 0, stream>>>(
        ffine, t1f, g1f, b1f, t2f, g2f, b2f, (float*)d_out, ffOrig);

    const long threadsScatter = (long)total * 32;
    const int blocksScatter = (int)((threadsScatter + 255) / 256);
    scatter_add_kernel<<<blocksScatter, 256, 0, stream>>>(
        fcBuf, ffOrig, nrow, ncol, (float*)d_out, total, Lc * N, N);
}
